// GCN_29429115912786
// MI455X (gfx1250) — matrix operand order repaired and checked by reading, not compiled, not run on hardware
//
#include <hip/hip_runtime.h>

typedef __attribute__((ext_vector_type(16))) _Float16 v16h;
typedef __attribute__((ext_vector_type(8)))  _Float16 v8h;
typedef __attribute__((ext_vector_type(8)))  float    v8f;

#define N_NODES   50000
#define N_EDGES   1600000
#define HID       128
#define OUT_FEATS 64

// ---------------------------------------------------------------------------
// Degree accumulation: deg_out[src[e]] += 1, deg_in[dst[e]] += 1
// ---------------------------------------------------------------------------
__global__ void gcn_deg_kernel(const int* __restrict__ src, const int* __restrict__ dst,
                               float* __restrict__ deg_out, float* __restrict__ deg_in,
                               int nE) {
  int e = blockIdx.x * blockDim.x + threadIdx.x;
  if (e < nE) {
    atomicAdd(&deg_out[src[e]], 1.0f);
    atomicAdd(&deg_in[dst[e]], 1.0f);
  }
}

// deg -> rsqrt(max(deg,1)) in place, for both arrays
__global__ void gcn_norm_kernel(float* __restrict__ ns, float* __restrict__ nd, int n) {
  int i = blockIdx.x * blockDim.x + threadIdx.x;
  if (i < n) {
    ns[i] = rsqrtf(fmaxf(ns[i], 1.0f));
    nd[i] = rsqrtf(fmaxf(nd[i], 1.0f));
  }
}

// ---------------------------------------------------------------------------
// One-time per layer: Wt[n][k] = (f16) W[k][n]  (transposed f16 copy, 32 KB).
// Puts the B operand in exactly the WMMA lane layout (lane=N, K-contiguous).
// ---------------------------------------------------------------------------
__global__ void gcn_wcvt_kernel(const float* __restrict__ W, _Float16* __restrict__ Wt) {
  int c = blockIdx.x;     // output column N  (0..127)
  int k = threadIdx.x;    // K               (0..127)
  Wt[(size_t)c * HID + k] = (_Float16)W[(size_t)k * HID + c];
}

// A16[row][k] = (f16)(features[row][k] * norm_src[row])   (layer-1 A operand)
__global__ void gcn_featcvt_kernel(const float* __restrict__ F, const float* __restrict__ ns,
                                   _Float16* __restrict__ A16) {
  int row = blockIdx.x;
  int c   = threadIdx.x;
  A16[(size_t)row * HID + c] = (_Float16)(F[(size_t)row * HID + c] * ns[row]);
}

// ---------------------------------------------------------------------------
// M = A16 @ Wt^T (both operands pre-scaled/converted f16)  [N x 128]x[128 x 128]
// One wave32 per 16x16 output tile; K=128 via 4x v_wmma_f32_16x16x32_f16.
// A layout (16-bit A 16x32, k order used here): lane = M (mod 16);
//   elems j=0..15 -> K = k0 + 16*hi + j
//   -> two contiguous 16-byte v8h loads per K-step.
// B layout (32x16): lane = N (mod 16); elems j=0..15 -> K = k0 + 16*hi + j
//   -> one contiguous v16h load from Wt[N][K].
// C/D layout (f32 16x16): vgpr r -> row = r + 8*hi, col = lane&15
// ---------------------------------------------------------------------------
__global__ __launch_bounds__(256) void gcn_gemm_wmma_kernel(
    const _Float16* __restrict__ A16, const _Float16* __restrict__ Wt,
    float* __restrict__ M) {
  const int wave  = threadIdx.x >> 5;
  const int lane  = threadIdx.x & 31;
  const int tile  = blockIdx.x * 8 + wave;     // 25000 tiles total
  const int mTile = tile >> 3;                 // 0..3124
  const int nTile = tile & 7;                  // 0..7
  const int hi    = lane >> 4;                 // half-wave select
  const int l15   = lane & 15;

  const int rowA = mTile * 16 + l15;
  const int colB = nTile * 16 + l15;

  v8f acc = {};
#pragma unroll
  for (int k0 = 0; k0 < HID; k0 += 32) {
    // ---- A fragment: two 16B runs of K-contiguous f16
    const _Float16* pa = A16 + (size_t)rowA * HID + k0 + hi * 16;
    const v8h a0 = *(const v8h*)pa;
    const v8h a1 = *(const v8h*)(pa + 8);
    v16h a;
#pragma unroll
    for (int j = 0; j < 8; ++j) { a[j] = a0[j]; a[j + 8] = a1[j]; }

    // ---- B fragment: 16 K-contiguous f16 from the transposed copy
    const v16h b = *(const v16h*)(Wt + (size_t)colB * HID + k0 + hi * 16);

    acc = __builtin_amdgcn_wmma_f32_16x16x32_f16(
        /*neg_a=*/false, a, /*neg_b=*/false, b,
        /*c_mod=*/(short)0, acc, /*reuse_a=*/false, /*reuse_b=*/false);
  }

  float* out = M + (size_t)(mTile * 16 + hi * 8) * HID + nTile * 16 + l15;
#pragma unroll
  for (int r = 0; r < 8; ++r) out[(size_t)r * HID] = acc[r];
}

// ---------------------------------------------------------------------------
// agg[dst[e]] += M[src[e]]  — one wave per edge, lane handles 4 consecutive
// floats (float4 gather + 4 global_atomic_add_f32). Both M and agg are
// L2-resident (25.6 MB each vs 192 MB L2).
// ---------------------------------------------------------------------------
__global__ __launch_bounds__(256) void gcn_scatter_kernel(
    const float* __restrict__ M, const int* __restrict__ src,
    const int* __restrict__ dst, float* __restrict__ agg, int nE) {
  int gwave = (int)((blockIdx.x * blockDim.x + threadIdx.x) >> 5);
  int lane  = threadIdx.x & 31;
  if (gwave < nE) {
    int s = src[gwave];
    int d = dst[gwave];
    const float4 v = *(const float4*)(M + (size_t)s * HID + lane * 4);
    float* ap = agg + (size_t)d * HID + lane * 4;
    atomicAdd(ap + 0, v.x);
    atomicAdd(ap + 1, v.y);
    atomicAdd(ap + 2, v.z);
    atomicAdd(ap + 3, v.w);
  }
}

// Layers 1-2 epilogue: next layer's A operand, pre-scaled + f16 in one pass:
// A16[row][c] = (f16)( relu(agg*nd[row] + b[c]) * ns[row] )
__global__ void gcn_relu_sc16_kernel(const float* __restrict__ agg, const float* __restrict__ nd,
                                     const float* __restrict__ ns, const float* __restrict__ bias,
                                     _Float16* __restrict__ A16) {
  int row = blockIdx.x;
  int c   = threadIdx.x;
  float v = fmaxf(agg[(size_t)row * HID + c] * nd[row] + bias[c], 0.0f);
  A16[(size_t)row * HID + c] = (_Float16)(v * ns[row]);
}

// Layer-3 epilogue: plain f32 relu (feeds the colsum), in-place safe
__global__ void gcn_relu_kernel(const float* __restrict__ agg, const float* __restrict__ nd,
                                const float* __restrict__ bias, float* __restrict__ Hout) {
  int row = blockIdx.x;
  int c   = threadIdx.x;
  float v = agg[(size_t)row * HID + c] * nd[row] + bias[c];
  Hout[(size_t)row * HID + c] = fmaxf(v, 0.0f);
}

// column sums of H (N x 128) -> cs[128]; each block reduces 256 rows locally
__global__ void gcn_colsum_kernel(const float* __restrict__ H, float* __restrict__ cs, int n) {
  int c  = threadIdx.x;            // 0..127
  int r0 = blockIdx.x * 256;
  int r1 = r0 + 256; if (r1 > n) r1 = n;
  float s = 0.0f;
  for (int r = r0; r < r1; ++r) s += H[(size_t)r * HID + c];
  atomicAdd(&cs[c], s);
}

// out[j] = (1/N) * sum_k cs[k]*fcW[k][j] + fcb[j]
__global__ void gcn_final_kernel(const float* __restrict__ cs, const float* __restrict__ fcW,
                                 const float* __restrict__ fcb, float* __restrict__ out) {
  int j = threadIdx.x;             // 0..63
  float s = 0.0f;
#pragma unroll 4
  for (int k = 0; k < HID; ++k) s += cs[k] * fcW[(size_t)k * OUT_FEATS + j];
  out[j] = s * (1.0f / (float)N_NODES) + fcb[j];
}

// ---------------------------------------------------------------------------
extern "C" void kernel_launch(void* const* d_in, const int* in_sizes, int n_in,
                              void* d_out, int out_size, void* d_ws, size_t ws_size,
                              hipStream_t stream) {
  const float* features = (const float*)d_in[0];
  const int*   src      = (const int*)d_in[1];
  const int*   dst      = (const int*)d_in[2];
  const float* W1 = (const float*)d_in[3];  const float* b1 = (const float*)d_in[4];
  const float* W2 = (const float*)d_in[5];  const float* b2 = (const float*)d_in[6];
  const float* W3 = (const float*)d_in[7];  const float* b3 = (const float*)d_in[8];
  const float* fcW = (const float*)d_in[9]; const float* fcb = (const float*)d_in[10];
  float* out = (float*)d_out;

  const size_t NF  = (size_t)N_NODES * HID * sizeof(float);     // 25.6 MB
  const size_t NF2 = (size_t)N_NODES * HID * sizeof(_Float16);  // 12.8 MB
  char* ws = (char*)d_ws;
  float*     Mbuf = (float*)ws;     ws += NF;                   // messages (f32)
  float*     Hbuf = (float*)ws;     ws += NF;                   // agg / final hidden (f32)
  _Float16*  A16  = (_Float16*)ws;  ws += NF2;                  // pre-scaled f16 A operand
  float*     ns   = (float*)ws;     ws += (size_t)N_NODES * sizeof(float);
  float*     nd   = (float*)ws;     ws += (size_t)N_NODES * sizeof(float);
  float*     cs   = (float*)ws;     ws += HID * sizeof(float);
  _Float16*  Wt   = (_Float16*)ws;  ws += (size_t)HID * HID * sizeof(_Float16);  // 32 KB
  (void)in_sizes; (void)n_in; (void)out_size; (void)ws_size;

  const int gemmBlocks    = N_NODES / 16;                    // 3125 blocks x 8 waves
  const int scatterBlocks = (N_EDGES + 7) / 8;               // 1 wave per edge, 8 waves/block

  // degrees -> norms (ns, nd are contiguous)
  hipMemsetAsync(ns, 0, 2 * (size_t)N_NODES * sizeof(float), stream);
  gcn_deg_kernel<<<(N_EDGES + 255) / 256, 256, 0, stream>>>(src, dst, ns, nd, N_EDGES);
  gcn_norm_kernel<<<(N_NODES + 255) / 256, 256, 0, stream>>>(ns, nd, N_NODES);

  // ---- Layer 1: features -> Hbuf/A16
  gcn_featcvt_kernel<<<N_NODES, HID, 0, stream>>>(features, ns, A16);
  gcn_wcvt_kernel<<<HID, HID, 0, stream>>>(W1, Wt);
  gcn_gemm_wmma_kernel<<<gemmBlocks, 256, 0, stream>>>(A16, Wt, Mbuf);
  hipMemsetAsync(Hbuf, 0, NF, stream);
  gcn_scatter_kernel<<<scatterBlocks, 256, 0, stream>>>(Mbuf, src, dst, Hbuf, N_EDGES);
  gcn_relu_sc16_kernel<<<N_NODES, HID, 0, stream>>>(Hbuf, nd, ns, b1, A16);

  // ---- Layer 2 (A16 is dead once the GEMM consumes it; epilogue rewrites it)
  gcn_wcvt_kernel<<<HID, HID, 0, stream>>>(W2, Wt);
  gcn_gemm_wmma_kernel<<<gemmBlocks, 256, 0, stream>>>(A16, Wt, Mbuf);
  hipMemsetAsync(Hbuf, 0, NF, stream);
  gcn_scatter_kernel<<<scatterBlocks, 256, 0, stream>>>(Mbuf, src, dst, Hbuf, N_EDGES);
  gcn_relu_sc16_kernel<<<N_NODES, HID, 0, stream>>>(Hbuf, nd, ns, b2, A16);

  // ---- Layer 3 (keep f32 hidden state for the colsum)
  gcn_wcvt_kernel<<<HID, HID, 0, stream>>>(W3, Wt);
  gcn_gemm_wmma_kernel<<<gemmBlocks, 256, 0, stream>>>(A16, Wt, Mbuf);
  hipMemsetAsync(Hbuf, 0, NF, stream);
  gcn_scatter_kernel<<<scatterBlocks, 256, 0, stream>>>(Mbuf, src, dst, Hbuf, N_EDGES);
  gcn_relu_kernel<<<N_NODES, HID, 0, stream>>>(Hbuf, nd, b3, Hbuf);

  // ---- FC + mean, folded: out = (colsum(H)/N) @ fcW + fcb
  hipMemsetAsync(cs, 0, HID * sizeof(float), stream);
  gcn_colsum_kernel<<<(N_NODES + 255) / 256, HID, 0, stream>>>(Hbuf, cs, N_NODES);
  gcn_final_kernel<<<1, OUT_FEATS, 0, stream>>>(cs, fcW, fcb, out);
}
